// DeepSetTraversalModule_81664508166536
// MI455X (gfx1250) — matrix operand order repaired and checked by reading, not compiled, not run on hardware
//
#include <hip/hip_runtime.h>
#include <hip/hip_fp16.h>

typedef _Float16 v16h __attribute__((ext_vector_type(16)));
typedef _Float16 v8h  __attribute__((ext_vector_type(8)));
typedef _Float16 v4h  __attribute__((ext_vector_type(4)));
typedef float    v8f  __attribute__((ext_vector_type(8)));
typedef float    v4f  __attribute__((ext_vector_type(4)));

#define CH 16
#define HH 96
#define WW 96
#define SB 20
#define BB 8
#define BNEPS 1e-5f

// ---------------------------------------------------------------- zero
__global__ void k_zero(float* p, int n) {
  int i = blockIdx.x * blockDim.x + threadIdx.x;
  if (i < n) p[i] = 0.f;
}

// --------------------------------------------- conv3x3 16->16 pad1, WMMA
// One block = one (image n, output row yo). 6 waves, each computes a
// 16-pixel x 16-outch tile with 5x v_wmma_f32_16x16x32_f16.
// LDS layouts are channel-fastest so every fragment group of 8/16 halves is
// one/two contiguous 16B reads (ds_load_b128), with zero-padded tap 9 so the
// inner loop has no conditionals.
__global__ __launch_bounds__(192) void k_conv3x3_wmma(
    const float* __restrict__ x, const float* __restrict__ w,
    float* __restrict__ y, float* __restrict__ stats)
{
  __shared__ __align__(16) _Float16 s_in[4][WW + 2][CH];  // [row y-1..y+1, zero][x pad][ci]
  __shared__ __align__(16) _Float16 s_wB[10][CH][CH];     // [tap 0..8, zero][co][ci]
  __shared__ float s_stat[2 * CH];

  const int blk = blockIdx.x;
  const int n   = blk / HH;
  const int yo  = blk % HH;
  const int tid = threadIdx.x;

  // stage input rows: 4 channels per thread -> one ds_store_b64
  for (int i = tid; i < 3 * (WW + 2) * 4; i += 192) {
    int xc = i % (WW + 2);
    int t  = i / (WW + 2);
    int cg = t & 3;                      // channel group of 4
    int ro = t >> 2;                     // 0..2
    int yi = yo + ro - 1;
    int xi = xc - 1;
    v4h pk;
    #pragma unroll
    for (int e = 0; e < 4; ++e) {
      float v = 0.f;
      if (yi >= 0 && yi < HH && xi >= 0 && xi < WW)
        v = x[(((size_t)n * CH + cg * 4 + e) * HH + yi) * WW + xi];
      pk[e] = (_Float16)v;
    }
    *(v4h*)&s_in[ro][xc][cg * 4] = pk;
  }
  // zero row 3 (pad tap 9)
  for (int i = tid; i < (WW + 2) * 4; i += 192) {
    int xc = i % (WW + 2), cg = i / (WW + 2);
    v4h z = {};
    *(v4h*)&s_in[3][xc][cg * 4] = z;
  }
  // weights transposed to [tap][co][ci]; tap 9 zeroed
  for (int i = tid; i < 9 * CH * CH; i += 192) {
    int p = i % 9; int t = i / 9; int ci = t % CH; int co = t / CH;
    s_wB[p][co][ci] = (_Float16)w[(co * CH + ci) * 9 + p];
  }
  for (int i = tid; i < CH * CH; i += 192) {
    int ci = i % CH, co = i / CH;
    s_wB[9][co][ci] = (_Float16)0.f;
  }
  if (tid < 2 * CH) s_stat[tid] = 0.f;
  __syncthreads();

  const int lane = tid & 31;
  const int wv   = tid >> 5;             // 0..5
  const int x0   = wv * 16;
  const int hf   = lane >> 4;            // lane half
  const int mrow = lane & 15;            // A-matrix M (pixel in tile)
  const int col  = lane & 15;            // B/D column (out channel)
  const int xa   = x0 + mrow;

  v8f acc = {};
  #pragma unroll
  for (int i = 0; i < 5; ++i) {
    const int p0 = 2 * i, p1 = 2 * i + 1;       // p1==9 -> zero row/tap
    v8h alo = *(const v8h*)&s_in[p0 / 3][xa + p0 % 3][hf * 8];
    v8h ahi = *(const v8h*)&s_in[p1 / 3][xa + p1 % 3][hf * 8];
    const int pb = 2 * i + hf;                  // pb==9 -> zero weights
    v8h blo = *(const v8h*)&s_wB[p0][col][hf * 8];
    v8h bhi = *(const v8h*)&s_wB[p1][col][hf * 8];
    v16h a = __builtin_shufflevector(alo, ahi, 0,1,2,3,4,5,6,7,8,9,10,11,12,13,14,15);
    v16h b = __builtin_shufflevector(blo, bhi, 0,1,2,3,4,5,6,7,8,9,10,11,12,13,14,15);
    acc = __builtin_amdgcn_wmma_f32_16x16x32_f16(false, a, false, b,
                                                 (short)0, acc, false, false);
  }

  // D rows held by a lane are 8 consecutive x -> two global_store_b128
  size_t obase = (((size_t)n * CH + col) * HH + yo) * WW + x0 + hf * 8;
  v4f o0 = {acc[0], acc[1], acc[2], acc[3]};
  v4f o1 = {acc[4], acc[5], acc[6], acc[7]};
  *(v4f*)&y[obase]     = o0;
  *(v4f*)&y[obase + 4] = o1;

  float ls = 0.f, lq = 0.f;
  #pragma unroll
  for (int r = 0; r < 8; ++r) { float v = acc[r]; ls += v; lq += v * v; }
  atomicAdd(&s_stat[col], ls);
  atomicAdd(&s_stat[CH + col], lq);
  __syncthreads();
  if (tid < CH) {
    atomicAdd(&stats[tid], s_stat[tid]);
    atomicAdd(&stats[CH + tid], s_stat[CH + tid]);
  }
}

// --------------------------------------------- perm-equi 1x1 conv (K=32), WMMA
// A = [support(16ch) ; loo(16ch)] per pixel, B = pw. One WMMA per tile.
// BN stats per (s, co) over (B,H,W).
__global__ __launch_bounds__(192) void k_pe_wmma(
    const float* __restrict__ sup, const float* __restrict__ sx,
    const float* __restrict__ tot, const float* __restrict__ pw,
    float* __restrict__ pe, float* __restrict__ stats)
{
  __shared__ __align__(16) _Float16 s_a[WW][32];   // [pixel][k] (k fastest)
  __shared__ __align__(16) _Float16 s_b[CH][32];   // [co][k] == pw flat layout
  __shared__ float s_stat[2 * CH];

  const int blk = blockIdx.x;
  const int bs  = blk / HH;                  // b*SB + s
  const int yo  = blk % HH;
  const int b   = bs / SB;
  const int s   = bs % SB;
  const int tid = threadIdx.x;

  const size_t chw      = (size_t)CH * HH * WW;
  const size_t sup_base = (size_t)bs * chw + (size_t)yo * WW;
  const size_t tot_base = (size_t)b  * chw + (size_t)yo * WW;

  for (int i = tid; i < WW * 4; i += 192) {
    int xi = i % WW, cg = i / WW;            // cg: channel group of 4
    v4h p0, p1;
    #pragma unroll
    for (int e = 0; e < 4; ++e) {
      int ci = cg * 4 + e;
      size_t o = (size_t)ci * HH * WW + xi;
      p0[e] = (_Float16)sup[sup_base + o];
      p1[e] = (_Float16)(tot[tot_base + o] - sx[sup_base + o]);  // leave-one-out
    }
    *(v4h*)&s_a[xi][cg * 4]      = p0;
    *(v4h*)&s_a[xi][16 + cg * 4] = p1;
  }
  for (int i = tid; i < CH * 32; i += 192)
    ((_Float16*)s_b)[i] = (_Float16)pw[i];   // pw: [co][32], matches s_b
  if (tid < 2 * CH) s_stat[tid] = 0.f;
  __syncthreads();

  const int lane = tid & 31;
  const int wv   = tid >> 5;
  const int x0   = wv * 16;
  const int hf   = lane >> 4;
  const int mrow = lane & 15;
  const int col  = lane & 15;
  const int xa   = x0 + mrow;

  v8h alo = *(const v8h*)&s_a[xa][hf * 8];
  v8h ahi = *(const v8h*)&s_a[xa][16 + hf * 8];
  v8h blo = *(const v8h*)&s_b[col][hf * 8];
  v8h bhi = *(const v8h*)&s_b[col][16 + hf * 8];
  v16h a = __builtin_shufflevector(alo, ahi, 0,1,2,3,4,5,6,7,8,9,10,11,12,13,14,15);
  v16h bf = __builtin_shufflevector(blo, bhi, 0,1,2,3,4,5,6,7,8,9,10,11,12,13,14,15);
  v8f acc = {};
  acc = __builtin_amdgcn_wmma_f32_16x16x32_f16(false, a, false, bf,
                                               (short)0, acc, false, false);

  size_t obase = (size_t)bs * chw + ((size_t)col * HH + yo) * WW + x0 + hf * 8;
  v4f o0 = {acc[0], acc[1], acc[2], acc[3]};
  v4f o1 = {acc[4], acc[5], acc[6], acc[7]};
  *(v4f*)&pe[obase]     = o0;
  *(v4f*)&pe[obase + 4] = o1;

  float ls = 0.f, lq = 0.f;
  #pragma unroll
  for (int r = 0; r < 8; ++r) { float v = acc[r]; ls += v; lq += v * v; }
  atomicAdd(&s_stat[col], ls);
  atomicAdd(&s_stat[CH + col], lq);
  __syncthreads();
  if (tid < CH) {
    atomicAdd(&stats[s * CH + tid], s_stat[tid]);
    atomicAdd(&stats[SB * CH + s * CH + tid], s_stat[CH + tid]);
  }
}

// ---------------------------------------------------------------- BN finalize
__global__ void k_bn_finalize(const float* __restrict__ stats,
                              const float* __restrict__ g, const float* __restrict__ bt,
                              float* __restrict__ ss, int nchan, float inv_cnt) {
  int c = blockIdx.x * blockDim.x + threadIdx.x;
  if (c >= nchan) return;
  float mean = stats[c] * inv_cnt;
  float var  = stats[nchan + c] * inv_cnt - mean * mean;
  float sc   = g[c & 15] * rsqrtf(var + BNEPS);
  ss[c]         = sc;
  ss[nchan + c] = bt[c & 15] - mean * sc;
}

// ------------------------------------------------- maxpool2(relu(bn(x)))
__global__ void k_pool_bn_relu(const float* __restrict__ in, const float* __restrict__ ss,
                               float* __restrict__ out, long total) {
  long i = (long)blockIdx.x * blockDim.x + threadIdx.x;
  if (i >= total) return;
  int  xo = (int)(i % 48); long t = i / 48;
  int  yo = (int)(t % 48); t /= 48;
  int  c  = (int)(t % CH); long n = t / CH;
  float sc = ss[c], sh = ss[CH + c];
  size_t base = (((size_t)n * CH + c) * HH + 2 * yo) * WW + 2 * xo;
  float v0 = fmaxf(in[base]          * sc + sh, 0.f);
  float v1 = fmaxf(in[base + 1]      * sc + sh, 0.f);
  float v2 = fmaxf(in[base + WW]     * sc + sh, 0.f);
  float v3 = fmaxf(in[base + WW + 1] * sc + sh, 0.f);
  out[i] = fmaxf(fmaxf(v0, v1), fmaxf(v2, v3));
}

// ------------------------------------------------- relu(bn(x)) in place
__global__ void k_bn_relu_ip(float* __restrict__ x, const float* __restrict__ ss, long total) {
  long i = (long)blockIdx.x * blockDim.x + threadIdx.x;
  if (i >= total) return;
  int c = (int)((i / (HH * WW)) % CH);
  x[i] = fmaxf(x[i] * ss[c] + ss[CH + c], 0.f);
}

// ------------------------------------------------- total = sum_s sx
__global__ void k_sum_s(const float* __restrict__ sx, float* __restrict__ tot) {
  long i = (long)blockIdx.x * blockDim.x + threadIdx.x;
  const size_t chw = (size_t)CH * HH * WW;
  if (i >= (long)((size_t)BB * chw)) return;
  long b = i / (long)chw; long r = i % (long)chw;
  float acc = 0.f;
  for (int s = 0; s < SB; ++s) acc += sx[(size_t)(b * SB + s) * chw + r];
  tot[i] = acc;
}

// ------------------------------------------------- deep = mean_s relu(bn_s(pe))
__global__ void k_pe_mean(const float* __restrict__ pe, const float* __restrict__ ss,
                          float* __restrict__ deep) {
  long i = (long)blockIdx.x * blockDim.x + threadIdx.x;
  const size_t chw = (size_t)CH * HH * WW;
  if (i >= (long)((size_t)BB * chw)) return;
  long b = i / (long)chw; long r = i % (long)chw;
  int  c = (int)(r / (HH * WW));
  float acc = 0.f;
  for (int s = 0; s < SB; ++s) {
    int ch = s * CH + c;
    float v = pe[(size_t)(b * SB + s) * chw + r] * ss[ch] + ss[SB * CH + ch];
    acc += fmaxf(v, 0.f);
  }
  deep[i] = acc * (1.f / (float)SB);
}

// ------------------------------ proj = maxpool2(softmax_c(bn(convj)))
__global__ void k_softmax_pool(const float* __restrict__ cj, const float* __restrict__ ss,
                               float* __restrict__ proj) {
  int i = blockIdx.x * blockDim.x + threadIdx.x;
  if (i >= BB * 48 * 48) return;
  int xo = i % 48, yo = (i / 48) % 48, b = i / (48 * 48);
  float pm[CH];
  #pragma unroll
  for (int c = 0; c < CH; ++c) pm[c] = 0.f;
  for (int q = 0; q < 4; ++q) {
    int yi = yo * 2 + (q >> 1), xi = xo * 2 + (q & 1);
    float v[CH]; float mx = -3.4e38f;
    #pragma unroll
    for (int c = 0; c < CH; ++c) {
      v[c] = cj[(((size_t)b * CH + c) * HH + yi) * WW + xi] * ss[c] + ss[CH + c];
      mx = fmaxf(mx, v[c]);
    }
    float sum = 0.f;
    #pragma unroll
    for (int c = 0; c < CH; ++c) { v[c] = __expf(v[c] - mx); sum += v[c]; }
    float inv = 1.f / sum;
    #pragma unroll
    for (int c = 0; c < CH; ++c) pm[c] = fmaxf(pm[c], v[c] * inv);
  }
  #pragma unroll
  for (int c = 0; c < CH; ++c)
    proj[(((size_t)b * CH + c) * 48 + yo) * 48 + xo] = pm[c];
}

// ------------------------------ outputs: [anchor_out | support_out]
__global__ void k_final_mul(const float* __restrict__ axp, const float* __restrict__ srp,
                            const float* __restrict__ proj, float* __restrict__ out) {
  const size_t AP = (size_t)BB * CH * 48 * 48;
  const size_t SP = (size_t)BB * SB * CH * 48 * 48;
  size_t i = (size_t)blockIdx.x * blockDim.x + threadIdx.x;
  if (i < AP) {
    out[i] = axp[i] * proj[i];
  } else if (i < AP + SP) {
    size_t j = i - AP;
    size_t p = j % (48 * 48);
    size_t c = (j / (48 * 48)) % CH;
    size_t b = j / ((size_t)48 * 48 * CH * SB);
    out[i] = srp[j] * proj[(b * CH + c) * (48 * 48) + p];
  }
}

// ================================================================ launch
extern "C" void kernel_launch(void* const* d_in, const int* in_sizes, int n_in,
                              void* d_out, int out_size, void* d_ws, size_t ws_size,
                              hipStream_t stream) {
  (void)in_sizes; (void)n_in; (void)out_size; (void)ws_size;
  const float* anchor  = (const float*)d_in[0];
  const float* support = (const float*)d_in[1];
  const float* cw  = (const float*)d_in[2];
  const float* cg  = (const float*)d_in[4];
  const float* cbt = (const float*)d_in[5];
  const float* pw  = (const float*)d_in[6];
  const float* pg  = (const float*)d_in[8];
  const float* pbt = (const float*)d_in[9];
  const float* jw  = (const float*)d_in[10];
  const float* jg  = (const float*)d_in[12];
  const float* jbt = (const float*)d_in[13];
  const float* rw  = (const float*)d_in[14];
  const float* rg  = (const float*)d_in[16];
  const float* rbt = (const float*)d_in[17];
  float* out = (float*)d_out;

  const size_t SUP_FULL = (size_t)BB * SB * CH * HH * WW;   // 23.6M f32
  const size_t ANC_FULL = (size_t)BB * CH * HH * WW;        // 1.18M f32
  const size_t SUP_POOL = (size_t)BB * SB * CH * 48 * 48;   // 5.90M f32
  const size_t ANC_POOL = (size_t)BB * CH * 48 * 48;        // 0.29M f32

  float* ws   = (float*)d_ws;
  float* bufA = ws;                       // reshaper conv(support), later pe
  float* bufB = bufA + SUP_FULL;          // concentrator conv(support) -> sx
  float* srp  = bufB + SUP_FULL;          // pooled support
  float* anc  = srp  + SUP_POOL;          // reshaper conv(anchor)
  float* tot  = anc  + ANC_FULL;          // sum_s(sx)
  float* deep = tot  + ANC_FULL;
  float* cvj  = deep + ANC_FULL;          // projector conv
  float* axp  = cvj  + ANC_FULL;          // pooled anchor
  float* proj = axp  + ANC_POOL;
  float* st   = proj + ANC_POOL;          // stats + scale/shift (1536 f32)
  float* st_rs = st;        float* st_ra = st + 32;  float* st_c = st + 64;
  float* st_j  = st + 96;   float* st_pe = st + 128; // 640
  float* ss_rs = st + 768;  float* ss_ra = st + 800; float* ss_c = st + 832;
  float* ss_j  = st + 864;  float* ss_pe = st + 896; // 640

  const float invSup = 1.f / (float)((size_t)BB * SB * HH * WW);
  const float invAnc = 1.f / (float)((size_t)BB * HH * WW);

  hipLaunchKernelGGL(k_zero, dim3(6), dim3(256), 0, stream, st, 1536);

  // convs (bias dropped: constant bias cancels in training-mode BN)
  hipLaunchKernelGGL(k_conv3x3_wmma, dim3(BB * SB * HH), dim3(192), 0, stream,
                     support, rw, bufA, st_rs);
  hipLaunchKernelGGL(k_conv3x3_wmma, dim3(BB * HH), dim3(192), 0, stream,
                     anchor, rw, anc, st_ra);
  hipLaunchKernelGGL(k_conv3x3_wmma, dim3(BB * SB * HH), dim3(192), 0, stream,
                     support, cw, bufB, st_c);

  hipLaunchKernelGGL(k_bn_finalize, dim3(1), dim3(32), 0, stream, st_rs, rg, rbt, ss_rs, CH, invSup);
  hipLaunchKernelGGL(k_bn_finalize, dim3(1), dim3(32), 0, stream, st_ra, rg, rbt, ss_ra, CH, invAnc);
  hipLaunchKernelGGL(k_bn_finalize, dim3(1), dim3(32), 0, stream, st_c,  cg, cbt, ss_c,  CH, invSup);

  hipLaunchKernelGGL(k_pool_bn_relu, dim3((unsigned)((SUP_POOL + 255) / 256)), dim3(256), 0, stream,
                     bufA, ss_rs, srp, (long)SUP_POOL);
  hipLaunchKernelGGL(k_pool_bn_relu, dim3((unsigned)((ANC_POOL + 255) / 256)), dim3(256), 0, stream,
                     anc, ss_ra, axp, (long)ANC_POOL);

  hipLaunchKernelGGL(k_bn_relu_ip, dim3((unsigned)((SUP_FULL + 255) / 256)), dim3(256), 0, stream,
                     bufB, ss_c, (long)SUP_FULL);
  hipLaunchKernelGGL(k_sum_s, dim3((unsigned)((ANC_FULL + 255) / 256)), dim3(256), 0, stream,
                     bufB, tot);

  // perm-equi 1x1 conv over [support ; loo], reuse bufA for pe
  hipLaunchKernelGGL(k_pe_wmma, dim3(BB * SB * HH), dim3(192), 0, stream,
                     support, bufB, tot, pw, bufA, st_pe);
  hipLaunchKernelGGL(k_bn_finalize, dim3((SB * CH + 63) / 64), dim3(64), 0, stream,
                     st_pe, pg, pbt, ss_pe, SB * CH, invAnc);
  hipLaunchKernelGGL(k_pe_mean, dim3((unsigned)((ANC_FULL + 255) / 256)), dim3(256), 0, stream,
                     bufA, ss_pe, deep);

  // projector
  hipLaunchKernelGGL(k_conv3x3_wmma, dim3(BB * HH), dim3(192), 0, stream, deep, jw, cvj, st_j);
  hipLaunchKernelGGL(k_bn_finalize, dim3(1), dim3(32), 0, stream, st_j, jg, jbt, ss_j, CH, invAnc);
  hipLaunchKernelGGL(k_softmax_pool, dim3((BB * 48 * 48 + 255) / 256), dim3(256), 0, stream,
                     cvj, ss_j, proj);

  hipLaunchKernelGGL(k_final_mul, dim3((unsigned)((ANC_POOL + SUP_POOL + 255) / 256)), dim3(256),
                     0, stream, axp, srp, proj, out);
}
